// TransformerModel_20753281974812
// MI455X (gfx1250) — compile-verified, matrix operand order checked by reading, not run on hardware
//
#include <hip/hip_runtime.h>

#define TOK 8192   // bs*S = 16*512
#define DM  512
#define NH  4
#define DK  128
#define SEQ 512
#define NB  16

typedef __attribute__((ext_vector_type(16))) __bf16 bf16x16;
typedef __attribute__((ext_vector_type(8)))  __bf16 bf16x8;
typedef __attribute__((ext_vector_type(8)))  float  f32x8;

__device__ __forceinline__ unsigned short f2bf(float f) {
    union { float f; unsigned int u; } v; v.f = f;
    unsigned int u = v.u;
    u += 0x7fffu + ((u >> 16) & 1u);   // round-to-nearest-even
    return (unsigned short)(u >> 16);
}
__device__ __forceinline__ float bf2f(unsigned short h) {
    union { unsigned int u; float f; } v; v.u = ((unsigned int)h) << 16;
    return v.f;
}

// ---------------------------------------------------------------------------
// Generic bf16 WMMA GEMM:  C[M,N] (+)= scale * (A[M,K] @ B^T[N,K]) + bias[N]
// A: bf16 row-major, row stride lda.  B: bf16 stored [N,K], row stride ldb.
// Batched via grid.y: z -> (bb = z/Hdim, hh = z%Hdim), per-operand offsets.
// Each wave computes a 32x64 output tile: 8 f32 accumulators, 2 A-fragments
// sharing 4 B-fragments (8 v_wmma per 32-K step), software-pipelined with
// register double-buffering over 64-K super-steps so loads for step k+1 are
// in flight while WMMAs for step k execute.
// Requires: M % 32 == 0, N % 64 == 0, K % 64 == 0.
// ---------------------------------------------------------------------------
template<bool OUTBF, bool ACCUM, bool RELU>
__global__ __launch_bounds__(256) void gemm_bf16_wmma(
    const unsigned short* __restrict__ A, int lda,
    const unsigned short* __restrict__ B, int ldb,
    void* __restrict__ Cv, int ldc,
    const float* __restrict__ bias, float scale,
    int M, int N, int K,
    long aOffB, long aOffH, long bOffB, long bOffH,
    long cOffB, long cOffH, int Hdim)
{
    const int lane = threadIdx.x & 31;
    const int wave = threadIdx.x >> 5;
    const int hl   = lane >> 4;      // which K half-pattern this lane holds
    const int r    = lane & 15;      // row (A) / column (B) within tile

    const int tilesN = N >> 6;
    const int tilesM = M >> 5;
    const int tileId = blockIdx.x * 8 + wave;
    if (tileId >= tilesM * tilesN) return;
    const int tm = tileId / tilesN;
    const int tn = tileId % tilesN;

    const int z  = blockIdx.y;
    const int bb = z / Hdim, hh = z % Hdim;
    const unsigned short* Ab = A + (long)bb * aOffB + (long)hh * aOffH;
    const unsigned short* Bb = B + (long)bb * bOffB + (long)hh * bOffH;

    const unsigned short* aRow0 = Ab + (long)(tm * 32 + r) * lda + hl * 8;
    const unsigned short* aRow1 = aRow0 + (long)16 * lda;
    const unsigned short* bRow[4];
#pragma unroll
    for (int j = 0; j < 4; ++j)
        bRow[j] = Bb + (long)(tn * 64 + j * 16 + r) * ldb + hl * 8;

    union Frag { bf16x16 v; bf16x8 h[2]; };
    auto LD = [](Frag& f, const unsigned short* p) {
        f.h[0] = *(const bf16x8*)(p);
        f.h[1] = *(const bf16x8*)(p + 16);
    };

    f32x8 acc[8] = {};
    Frag a0A, a1A, bA[4];   // ping buffer
    Frag a0B, a1B, bB[4];   // pong buffer

    // prologue: fill ping with k = 0
    LD(a0A, aRow0); LD(a1A, aRow1);
#pragma unroll
    for (int j = 0; j < 4; ++j) LD(bA[j], bRow[j]);

    for (int k0 = 0; k0 < K; k0 += 64) {
        // prefetch pong (k0 + 32) while ping computes
        LD(a0B, aRow0 + k0 + 32); LD(a1B, aRow1 + k0 + 32);
#pragma unroll
        for (int j = 0; j < 4; ++j) LD(bB[j], bRow[j] + k0 + 32);
#pragma unroll
        for (int j = 0; j < 4; ++j) {
            acc[j]     = __builtin_amdgcn_wmma_f32_16x16x32_bf16(
                false, a0A.v, false, bA[j].v, (short)0, acc[j],     false, false);
            acc[4 + j] = __builtin_amdgcn_wmma_f32_16x16x32_bf16(
                false, a1A.v, false, bA[j].v, (short)0, acc[4 + j], false, false);
        }
        // prefetch next ping (k0 + 64) while pong computes
        if (k0 + 64 < K) {
            LD(a0A, aRow0 + k0 + 64); LD(a1A, aRow1 + k0 + 64);
#pragma unroll
            for (int j = 0; j < 4; ++j) LD(bA[j], bRow[j] + k0 + 64);
        }
#pragma unroll
        for (int j = 0; j < 4; ++j) {
            acc[j]     = __builtin_amdgcn_wmma_f32_16x16x32_bf16(
                false, a0B.v, false, bB[j].v, (short)0, acc[j],     false, false);
            acc[4 + j] = __builtin_amdgcn_wmma_f32_16x16x32_bf16(
                false, a1B.v, false, bB[j].v, (short)0, acc[4 + j], false, false);
        }
    }

    // C/D layout: VGPR i holds (M = i + 8*half, N = lane%16)
    if (OUTBF) {
        unsigned short* C = (unsigned short*)Cv + (long)bb * cOffB + (long)hh * cOffH;
#pragma unroll
        for (int j = 0; j < 4; ++j) {
            int col = tn * 64 + j * 16 + r;
            float bv = bias ? bias[col] : 0.f;
#pragma unroll
            for (int s = 0; s < 2; ++s) {
#pragma unroll
                for (int i = 0; i < 8; ++i) {
                    int row = tm * 32 + s * 16 + hl * 8 + i;
                    float val = acc[s * 4 + j][i] * scale + bv;
                    if (RELU) val = fmaxf(val, 0.f);
                    C[(long)row * ldc + col] = f2bf(val);
                }
            }
        }
    } else {
        float* C = (float*)Cv + (long)bb * cOffB + (long)hh * cOffH;
#pragma unroll
        for (int j = 0; j < 4; ++j) {
            int col = tn * 64 + j * 16 + r;
            float bv = bias ? bias[col] : 0.f;
#pragma unroll
            for (int s = 0; s < 2; ++s) {
#pragma unroll
                for (int i = 0; i < 8; ++i) {
                    int row = tm * 32 + s * 16 + hl * 8 + i;
                    float val = acc[s * 4 + j][i] * scale + bv;
                    if (RELU) val = fmaxf(val, 0.f);
                    long idx = (long)row * ldc + col;
                    if (ACCUM) C[idx] += val; else C[idx] = val;
                }
            }
        }
    }
}

// fp32 [K,N] -> bf16 transposed [N,K]; batched along grid.y
__global__ void wtrans(const float* __restrict__ src, unsigned short* __restrict__ dst,
                       int Kd, int Nd)
{
    long base = (long)blockIdx.y * Kd * Nd;
    long i = (long)blockIdx.x * blockDim.x + threadIdx.x;
    if (i >= (long)Kd * Nd) return;
    long k = i % Kd;
    long n = i / Kd;
    dst[base + n * Kd + k] = f2bf(src[base + k * Nd + n]);
}

// h = x @ in_w + in_b + cond @ cond_w + cond_b  (K = 18 and 89: scalar FMAs)
__global__ void input_proj(const float* __restrict__ x, const float* __restrict__ cond,
                           const float* __restrict__ in_w, const float* __restrict__ in_b,
                           const float* __restrict__ cond_w, const float* __restrict__ cond_b,
                           float* __restrict__ h)
{
    long idx = (long)blockIdx.x * blockDim.x + threadIdx.x;
    if (idx >= (long)TOK * DM) return;
    int n = (int)(idx & (DM - 1));
    long t = idx >> 9;
    float acc = in_b[n] + cond_b[n];
    const float* xr = x + t * 18;
#pragma unroll
    for (int k = 0; k < 18; ++k) acc += xr[k] * in_w[k * DM + n];
    const float* cr = cond + t * 89;
    for (int k = 0; k < 89; ++k) acc += cr[k] * cond_w[k * DM + n];
    h[idx] = acc;
}

// InstanceNorm over D=512 per token (biased var, eps=1e-5), output bf16.
// One wave32 per token: 16 values/lane + shfl_xor reduction.
__global__ void inorm_cvt(const float* __restrict__ x, unsigned short* __restrict__ y, int ntok)
{
    int tok  = (int)(((long)blockIdx.x * blockDim.x + threadIdx.x) >> 5);
    int lane = threadIdx.x & 31;
    if (tok >= ntok) return;
    const float* row = x + (long)tok * DM;
    float v[16], s = 0.f, s2 = 0.f;
#pragma unroll
    for (int i = 0; i < 16; ++i) {
        float t = row[lane + 32 * i];
        v[i] = t; s += t; s2 += t * t;
    }
#pragma unroll
    for (int off = 16; off >= 1; off >>= 1) {
        s  += __shfl_xor(s,  off, 32);
        s2 += __shfl_xor(s2, off, 32);
    }
    float mean = s * (1.f / DM);
    float var  = s2 * (1.f / DM) - mean * mean;
    float inv  = rsqrtf(var + 1e-5f);
    unsigned short* orow = y + (long)tok * DM;
#pragma unroll
    for (int i = 0; i < 16; ++i)
        orow[lane + 32 * i] = f2bf((v[i] - mean) * inv);
}

__global__ void cvt_f32_bf16(const float* __restrict__ a, unsigned short* __restrict__ b, long n)
{
    long i = (long)blockIdx.x * blockDim.x + threadIdx.x;
    if (i < n) b[i] = f2bf(a[i]);
}

// masked softmax over rows of [64, 512, 512] fp32 scores -> bf16 probs
__global__ void softmax_mask(const float* __restrict__ scores, const int* __restrict__ mask,
                             unsigned short* __restrict__ probs)
{
    long row = ((long)blockIdx.x * blockDim.x + threadIdx.x) >> 5;
    int lane = threadIdx.x & 31;
    if (row >= (long)NB * NH * SEQ) return;
    int zi = (int)(row >> 9);          // b*NH + h
    int i  = (int)(row & (SEQ - 1));
    int b  = zi >> 2;
    const float* srow = scores + row * SEQ;
    const int*   mrow = mask + ((long)b * SEQ + i) * SEQ;
    float v[16], mx = -3.0e38f;
#pragma unroll
    for (int t = 0; t < 16; ++t) {
        int j = lane + 32 * t;
        float s = (mrow[j] == 1) ? -1e9f : srow[j];
        v[t] = s; mx = fmaxf(mx, s);
    }
#pragma unroll
    for (int off = 16; off >= 1; off >>= 1) mx = fmaxf(mx, __shfl_xor(mx, off, 32));
    float sum = 0.f;
#pragma unroll
    for (int t = 0; t < 16; ++t) { float e = __expf(v[t] - mx); v[t] = e; sum += e; }
#pragma unroll
    for (int off = 16; off >= 1; off >>= 1) sum += __shfl_xor(sum, off, 32);
    float inv = 1.f / sum;
    unsigned short* prow = probs + row * SEQ;
#pragma unroll
    for (int t = 0; t < 16; ++t) prow[lane + 32 * t] = f2bf(v[t] * inv);
}

// v [token, D] bf16 -> vt [b*NH+h][DK][SEQ] bf16 (per-head transpose for PV GEMM)
__global__ void vtrans(const unsigned short* __restrict__ v, unsigned short* __restrict__ vt)
{
    long i = (long)blockIdx.x * blockDim.x + threadIdx.x;
    if (i >= (long)NB * NH * DK * SEQ) return;
    int j = (int)(i & (SEQ - 1));
    int d = (int)((i >> 9) & (DK - 1));
    int z = (int)(i >> 16);
    int b = z >> 2, h = z & 3;
    vt[i] = v[((long)(b * SEQ + j)) * DM + h * DK + d];
}

// final head: out[t, 0..1] = relu-chain result @ out3_w + out3_b
__global__ void head_out3(const unsigned short* __restrict__ a, const float* __restrict__ w,
                          const float* __restrict__ b, float* __restrict__ out)
{
    long idx = (long)blockIdx.x * blockDim.x + threadIdx.x;
    if (idx >= (long)TOK * 2) return;
    int o = (int)(idx & 1);
    long t = idx >> 1;
    float acc = b[o];
    const unsigned short* ar = a + t * 256;
    for (int k = 0; k < 256; ++k) acc += bf2f(ar[k]) * w[k * 2 + o];
    out[idx] = acc;
}

extern "C" void kernel_launch(void* const* d_in, const int* in_sizes, int n_in,
                              void* d_out, int out_size, void* d_ws, size_t ws_size,
                              hipStream_t stream)
{
    (void)in_sizes; (void)n_in; (void)out_size; (void)ws_size;
    const float* x      = (const float*)d_in[0];
    const float* cond   = (const float*)d_in[1];
    const int*   door_m = (const int*)d_in[2];
    const int*   self_m = (const int*)d_in[3];
    const int*   gen_m  = (const int*)d_in[4];
    const float* attn_w = (const float*)d_in[5];
    const float* attn_b = (const float*)d_in[6];
    const float* ff_w1  = (const float*)d_in[7];
    const float* ff_b1  = (const float*)d_in[8];
    const float* ff_w2  = (const float*)d_in[9];
    const float* ff_b2  = (const float*)d_in[10];
    const float* in_w   = (const float*)d_in[11];
    const float* in_b   = (const float*)d_in[12];
    const float* cond_w = (const float*)d_in[13];
    const float* cond_b = (const float*)d_in[14];
    const float* out1_w = (const float*)d_in[15];
    const float* out1_b = (const float*)d_in[16];
    const float* out2_w = (const float*)d_in[17];
    const float* out2_b = (const float*)d_in[18];
    const float* out3_w = (const float*)d_in[19];
    const float* out3_b = (const float*)d_in[20];

    size_t off = 0;
    auto alloc = [&](size_t bytes) -> void* {
        off = (off + 255) & ~(size_t)255;
        void* p = (char*)d_ws + off;
        off += bytes;
        return p;
    };
    unsigned short* wAttn = (unsigned short*)alloc(48UL * 512 * 512 * 2);
    unsigned short* wFf1  = (unsigned short*)alloc(4UL * 1024 * 512 * 2);
    unsigned short* wFf2  = (unsigned short*)alloc(4UL * 512 * 1024 * 2);
    unsigned short* wOut1 = (unsigned short*)alloc(512UL * 512 * 2);
    unsigned short* wOut2 = (unsigned short*)alloc(256UL * 512 * 2);
    float*          hbuf  = (float*)alloc((size_t)TOK * DM * 4);
    unsigned short* h2    = (unsigned short*)alloc((size_t)TOK * DM * 2);
    unsigned short* qb    = (unsigned short*)alloc((size_t)TOK * DM * 2);
    unsigned short* kb    = (unsigned short*)alloc((size_t)TOK * DM * 2);
    unsigned short* vb    = (unsigned short*)alloc((size_t)TOK * DM * 2);
    unsigned short* vt    = (unsigned short*)alloc((size_t)NB * NH * DK * SEQ * 2);
    float*          sc    = (float*)alloc((size_t)NB * NH * SEQ * SEQ * 4);
    unsigned short* pb    = (unsigned short*)alloc((size_t)NB * NH * SEQ * SEQ * 2);
    unsigned short* ct    = (unsigned short*)alloc((size_t)TOK * DM * 2);
    unsigned short* ffb   = (unsigned short*)alloc((size_t)TOK * 1024 * 2);
    unsigned short* o1    = (unsigned short*)alloc((size_t)TOK * DM * 2);
    unsigned short* o2    = (unsigned short*)alloc((size_t)TOK * 256 * 2);

    // ---- weight convert + transpose (fp32 [K,N] -> bf16 [N,K]) ----
    wtrans<<<dim3((512 * 512 + 255) / 256, 48), 256, 0, stream>>>(attn_w, wAttn, 512, 512);
    wtrans<<<dim3((512 * 1024 + 255) / 256, 4), 256, 0, stream>>>(ff_w1, wFf1, 512, 1024);
    wtrans<<<dim3((1024 * 512 + 255) / 256, 4), 256, 0, stream>>>(ff_w2, wFf2, 1024, 512);
    wtrans<<<dim3((512 * 512 + 255) / 256, 1), 256, 0, stream>>>(out1_w, wOut1, 512, 512);
    wtrans<<<dim3((512 * 256 + 255) / 256, 1), 256, 0, stream>>>(out2_w, wOut2, 512, 256);

    // ---- input + cond projection ----
    input_proj<<<(TOK * DM) / 256, 256, 0, stream>>>(x, cond, in_w, in_b, cond_w, cond_b, hbuf);

    const int* masks[3] = {door_m, self_m, gen_m};
    const float inv_sqrt_dk = 0.08838834764831845f;  // 1/sqrt(128)

    // grid helper: 32x64 tiles, 8 waves (tiles) per 256-thread block
    #define GEMM_GRID(Mv, Nv, Zv) dim3((unsigned)(((Mv) / 32) * ((Nv) / 64) / 8), (unsigned)(Zv))

    for (int l = 0; l < 4; ++l) {
        // pre-norm shared by all three attention branches
        inorm_cvt<<<(TOK * 32) / 256, 256, 0, stream>>>(hbuf, h2, TOK);

        for (int br = 0; br < 3; ++br) {
            int wi = (l * 3 + br) * 4;
            // Q, K, V projections: [8192,512] @ [512,512]^T + bias -> bf16
            gemm_bf16_wmma<true, false, false><<<GEMM_GRID(TOK, 512, 1), 256, 0, stream>>>(
                h2, 512, wAttn + (size_t)(wi + 0) * 262144, 512, qb, 512,
                attn_b + (size_t)(wi + 0) * 512, 1.f, TOK, 512, 512,
                0, 0, 0, 0, 0, 0, 1);
            gemm_bf16_wmma<true, false, false><<<GEMM_GRID(TOK, 512, 1), 256, 0, stream>>>(
                h2, 512, wAttn + (size_t)(wi + 1) * 262144, 512, kb, 512,
                attn_b + (size_t)(wi + 1) * 512, 1.f, TOK, 512, 512,
                0, 0, 0, 0, 0, 0, 1);
            gemm_bf16_wmma<true, false, false><<<GEMM_GRID(TOK, 512, 1), 256, 0, stream>>>(
                h2, 512, wAttn + (size_t)(wi + 2) * 262144, 512, vb, 512,
                attn_b + (size_t)(wi + 2) * 512, 1.f, TOK, 512, 512,
                0, 0, 0, 0, 0, 0, 1);
            // per-head transpose of V for the PV GEMM
            vtrans<<<(NB * NH * DK * SEQ) / 256, 256, 0, stream>>>(vb, vt);
            // scores = (Q @ K^T) / sqrt(dk), batched over 64 (b,h)
            gemm_bf16_wmma<false, false, false><<<GEMM_GRID(512, 512, 64), 256, 0, stream>>>(
                qb, 512, kb, 512, sc, 512, nullptr, inv_sqrt_dk,
                512, 512, 128,
                262144, 128, 262144, 128, 1048576, 262144, 4);
            // masked softmax -> bf16 probs
            softmax_mask<<<(NB * NH * SEQ * 32) / 256, 256, 0, stream>>>(sc, masks[br], pb);
            // out = P @ V  (writes directly into concat [token, h*128+d] layout)
            gemm_bf16_wmma<true, false, false><<<GEMM_GRID(512, 128, 64), 256, 0, stream>>>(
                pb, 512, vt, 512, ct, 512, nullptr, 1.f,
                512, 128, 512,
                1048576, 262144, 262144, 65536, 262144, 128, 4);
            // out-projection, accumulated into residual stream hbuf (fp32)
            gemm_bf16_wmma<false, true, false><<<GEMM_GRID(TOK, 512, 1), 256, 0, stream>>>(
                ct, 512, wAttn + (size_t)(wi + 3) * 262144, 512, hbuf, 512,
                attn_b + (size_t)(wi + 3) * 512, 1.f, TOK, 512, 512,
                0, 0, 0, 0, 0, 0, 1);
        }

        // FFN pre-norm
        inorm_cvt<<<(TOK * 32) / 256, 256, 0, stream>>>(hbuf, h2, TOK);
        // FF1 + ReLU -> bf16 [8192,1024]
        gemm_bf16_wmma<true, false, true><<<GEMM_GRID(TOK, 1024, 1), 256, 0, stream>>>(
            h2, 512, wFf1 + (size_t)l * 524288, 512, ffb, 1024,
            ff_b1 + (size_t)l * 1024, 1.f, TOK, 1024, 512,
            0, 0, 0, 0, 0, 0, 1);
        // FF2, accumulated into hbuf (residual)
        gemm_bf16_wmma<false, true, false><<<GEMM_GRID(TOK, 512, 1), 256, 0, stream>>>(
            ffb, 1024, wFf2 + (size_t)l * 524288, 1024, hbuf, 512,
            ff_b2 + (size_t)l * 512, 1.f, TOK, 512, 1024,
            0, 0, 0, 0, 0, 0, 1);
    }

    // ---- output head ----
    cvt_f32_bf16<<<(TOK * DM) / 256, 256, 0, stream>>>(hbuf, h2, (long)TOK * DM);
    gemm_bf16_wmma<true, false, true><<<GEMM_GRID(TOK, 512, 1), 256, 0, stream>>>(
        h2, 512, wOut1, 512, o1, 512, out1_b, 1.f, TOK, 512, 512,
        0, 0, 0, 0, 0, 0, 1);
    gemm_bf16_wmma<true, false, true><<<GEMM_GRID(TOK, 256, 1), 256, 0, stream>>>(
        o1, 512, wOut2, 512, o2, 256, out2_b, 1.f, TOK, 256, 512,
        0, 0, 0, 0, 0, 0, 1);
    head_out3<<<(TOK * 2 + 255) / 256, 256, 0, stream>>>(o2, out3_w, out3_b, (float*)d_out);

    #undef GEMM_GRID
}
